// SparseAttention_59399397704373
// MI455X (gfx1250) — compile-verified, matrix operand order checked by reading, not run on hardware
//
#include <hip/hip_runtime.h>

#if __has_builtin(__builtin_amdgcn_tensor_load_to_lds)
#define USE_TDM 1
#else
#define USE_TDM 0
#endif

// ---- CDNA5 (gfx1250) types ------------------------------------------------
typedef __attribute__((ext_vector_type(16))) __bf16 v16bf;
typedef __attribute__((ext_vector_type(8)))  __bf16 v8bf;
typedef __attribute__((ext_vector_type(8)))  float  v8f;
typedef __attribute__((ext_vector_type(4)))  unsigned int u32x4;
typedef __attribute__((ext_vector_type(8)))  int i32x8;
typedef __attribute__((ext_vector_type(4)))  int i32x4;

// ---- problem constants ----------------------------------------------------
constexpr int B_  = 4;
constexpr int S_  = 4096;
constexpr int D_  = 128;
constexpr int BS  = 64;        // mask block size
constexpr int NB  = S_ / BS;   // 64 blocks
constexpr int W_  = NB / 2;    // band half-width = 32 blocks

// LDS strides (bf16 elements): multiples of 8 (16B-aligned rows), chosen so
// 16-lane stripes of 16B reads tile all 64 LDS banks.
constexpr int KSTR = 136;      // K tile   [64 rows ][D=128] row-major
constexpr int VSTR = 72;       // V tile   [D=128  ][64 rows] transposed
constexpr int PSTR = 72;       // P buffer [16 rows][64 cols] per wave

__device__ inline __bf16 f2bf(float x) { return (__bf16)x; }

// 16x32 A-frag / 32x16 B-frag bf16 lane image (ISA 7.12.2): lanes 0-15 hold
// k in {0..7}U{16..23}, lanes 16-31 hold {8..15}U{24..31}; caller pre-offsets
// p by laneHalf*8.
__device__ inline v16bf make_frag(const __bf16* p) {
  v8bf lo = *(const v8bf*)(p);
  v8bf hi = *(const v8bf*)(p + 16);
  v16bf f;
#pragma unroll
  for (int i = 0; i < 8; ++i) { f[i] = lo[i]; f[8 + i] = hi[i]; }
  return f;
}

__device__ inline v16bf make_frag_f32(const float* p) {
  v16bf f;
#pragma unroll
  for (int i = 0; i < 8; ++i) { f[i] = f2bf(p[i]); f[8 + i] = f2bf(p[16 + i]); }
  return f;
}

#if USE_TDM
// Issue a TDM 2D tile load (f32 elements) global -> LDS.  D# layout per
// CDNA5 ISA §8.3-8.5: group0 = {count=1, lds_addr, global_addr, type=2};
// group1 = {data_size=2(4B), tensor_dim0/1, tile_dim0/1, dim0 stride}.
__device__ inline void tdm_load_2d_f32(unsigned lds_byte_addr, const float* gptr,
                                       unsigned rows, unsigned cols,
                                       unsigned row_stride_elems) {
  const unsigned long long ga = (unsigned long long)(uintptr_t)gptr;
  u32x4 g0;
  g0[0] = 1u;                                                  // count=1
  g0[1] = lds_byte_addr;                                       // lds_addr
  g0[2] = (unsigned)ga;                                        // global_addr lo
  g0[3] = (unsigned)((ga >> 32) & 0x01FFFFFFu) | (2u << 30);   // addr hi | type=2
  i32x8 g1;
  g1[0] = (int)(2u << 16);                                     // data_size = 4B
  g1[1] = (int)((cols & 0xFFFFu) << 16);                       // tensor_dim0 lo
  g1[2] = (int)((cols >> 16) | ((rows & 0xFFFFu) << 16));      // dim0 hi | dim1 lo
  g1[3] = (int)((rows >> 16) | ((cols & 0xFFFFu) << 16));      // dim1 hi | tile_dim0
  g1[4] = (int)(rows & 0xFFFFu);                               // tile_dim1, tile_dim2=0
  g1[5] = (int)row_stride_elems;                               // tensor_dim0_stride lo
  g1[6] = 0;
  g1[7] = 0;
  const i32x4 z4 = {0, 0, 0, 0};
#if __has_include(<hip/amd_detail/amd_gfx1250_TDM.h>)
  const i32x8 z8 = {0, 0, 0, 0, 0, 0, 0, 0};
  __builtin_amdgcn_tensor_load_to_lds(g0, g1, z4, z4, z8, 0);  // 6-arg toolchain
#else
  __builtin_amdgcn_tensor_load_to_lds(g0, g1, z4, z4, 0);      // 5-arg toolchain
#endif
}
#endif

// ---- kernel 1: column sums of V, Vsum[b][d] = sum_s V[b][s][d] ------------
__global__ void vsum_kernel(const float* __restrict__ V, float* __restrict__ vsum) {
  const int b = blockIdx.x;
  const int d = threadIdx.x;
  const float* vp = V + (size_t)b * S_ * D_ + d;
  float s = 0.f;
  for (int r = 0; r < S_; ++r) s += vp[(size_t)r * D_];
  vsum[b * D_ + d] = s;
}

// ---- kernel 2: band flash-attention, TDM-pipelined staging ----------------
__global__ __launch_bounds__(128, 1)
void SparseAttention_59399397704373_kernel(const float* __restrict__ Qg,
                                           const float* __restrict__ Kg,
                                           const float* __restrict__ Vg,
                                           const float* __restrict__ vsum,
                                           float* __restrict__ out) {
#if USE_TDM
  __shared__ float  Fk [BS * D_];          // 32 KB raw f32 K staging (TDM dest)
  __shared__ float  Fv [BS * D_];          // 32 KB raw f32 V staging (TDM dest)
#endif
  __shared__ __bf16 Kl [BS * KSTR];        // 17 KB
  __shared__ __bf16 VTl[D_ * VSTR];        // 18 KB (V transposed)
  __shared__ __bf16 Pl [4 * 16 * PSTR];    //  9 KB (per-wave P staging)
  __shared__ float  Vbp[4][D_];            // partial band column-sums of V

  const int ib   = blockIdx.x;             // query block 0..63
  const int b    = blockIdx.y;             // batch
  const int tid  = threadIdx.x;
  const int wave = tid >> 5;
  const int lane = tid & 31;
  const int lm   = lane & 15;
  const int laneHalf = lane >> 4;

  const int lo    = max(0, ib - W_);
  const int hi    = min(NB, ib + W_);
  const int n_out = S_ - (hi - lo) * BS;   // zero-score out-of-band columns

  // ---- load Q A-frags (wave handles rows q0..q0+15, full D=128) ----------
  const int q0 = ib * BS + wave * 16;
  const float* qrow = Qg + ((size_t)b * S_ + q0 + lm) * D_;
  v16bf qf[4];
#pragma unroll
  for (int kk = 0; kk < 4; ++kk)
    qf[kk] = make_frag_f32(qrow + kk * 32 + laneHalf * 8);

  v8f   acc[8] = {};
  float m_i[8], l_i[8];
#pragma unroll
  for (int v = 0; v < 8; ++v) { m_i[v] = -1.0e30f; l_i[v] = 0.f; }
  float vb4[4] = {0.f, 0.f, 0.f, 0.f};     // band V colsum, 4 dims per thread

#if USE_TDM
  const unsigned fk_lds = (unsigned)(uintptr_t)(&Fk[0]);
  const unsigned fv_lds = (unsigned)(uintptr_t)(&Fv[0]);
  // Pre-issue TDM loads for the first band block (wave 0 drives the TDM).
  if (wave == 0) {
    const size_t g0off = ((size_t)b * S_ + (size_t)lo * BS) * D_;
    tdm_load_2d_f32(fk_lds, Kg + g0off, BS, D_, D_);
    tdm_load_2d_f32(fv_lds, Vg + g0off, BS, D_, D_);
  }
#endif

  for (int j = lo; j < hi; ++j) {
#if USE_TDM
    // ---- consume TDM staging for block j, convert f32 -> bf16 tiles ------
    if (wave == 0) __builtin_amdgcn_s_wait_tensorcnt(0);
    __syncthreads();                       // Fk/Fv(j) visible to all waves
#pragma unroll 4
    for (int t = tid; t < BS * (D_ / 4); t += 128) {
      const int r  = t >> 5;
      const int d4 = (t & 31) << 2;
      const float4 kv = *reinterpret_cast<const float4*>(&Fk[r * D_ + d4]);
      const float4 vv = *reinterpret_cast<const float4*>(&Fv[r * D_ + d4]);
      __bf16* kd = &Kl[r * KSTR + d4];
      kd[0] = f2bf(kv.x); kd[1] = f2bf(kv.y); kd[2] = f2bf(kv.z); kd[3] = f2bf(kv.w);
      VTl[(d4 + 0) * VSTR + r] = f2bf(vv.x);
      VTl[(d4 + 1) * VSTR + r] = f2bf(vv.y);
      VTl[(d4 + 2) * VSTR + r] = f2bf(vv.z);
      VTl[(d4 + 3) * VSTR + r] = f2bf(vv.w);
      vb4[0] += vv.x; vb4[1] += vv.y; vb4[2] += vv.z; vb4[3] += vv.w;
    }
    __syncthreads();                       // bf16 tiles ready; Fk/Fv free
    // ---- prefetch block j+1 via TDM while we compute block j -------------
    if (j + 1 < hi && wave == 0) {
      const size_t gn = ((size_t)b * S_ + (size_t)(j + 1) * BS) * D_;
      tdm_load_2d_f32(fk_lds, Kg + gn, BS, D_, D_);
      tdm_load_2d_f32(fv_lds, Vg + gn, BS, D_, D_);
    }
#else
    // ---- synchronous staging fallback ------------------------------------
    if (j + 1 < hi) {
      const size_t gn = ((size_t)b * S_ + (size_t)(j + 1) * BS) * D_;
      __builtin_prefetch(Kg + gn + tid * 16, 0, 0);
      __builtin_prefetch(Vg + gn + tid * 16, 0, 0);
    }
#pragma unroll 4
    for (int t = tid; t < BS * (D_ / 4); t += 128) {
      const int r  = t >> 5;
      const int d4 = (t & 31) << 2;
      const size_t g = ((size_t)b * S_ + (size_t)j * BS + r) * D_ + d4;
      const float4 kv = *reinterpret_cast<const float4*>(Kg + g);
      const float4 vv = *reinterpret_cast<const float4*>(Vg + g);
      __bf16* kd = &Kl[r * KSTR + d4];
      kd[0] = f2bf(kv.x); kd[1] = f2bf(kv.y); kd[2] = f2bf(kv.z); kd[3] = f2bf(kv.w);
      VTl[(d4 + 0) * VSTR + r] = f2bf(vv.x);
      VTl[(d4 + 1) * VSTR + r] = f2bf(vv.y);
      VTl[(d4 + 2) * VSTR + r] = f2bf(vv.z);
      VTl[(d4 + 3) * VSTR + r] = f2bf(vv.w);
      vb4[0] += vv.x; vb4[1] += vv.y; vb4[2] += vv.z; vb4[3] += vv.w;
    }
    __syncthreads();
#endif

    // ---- S = Q . K^T  (16x64, 4 N-tiles, K-dim 128 in 4 steps of 32) -----
    v8f sfrag[4];
#pragma unroll
    for (int nt = 0; nt < 4; ++nt) {
      v8f c = {};
#pragma unroll
      for (int kk = 0; kk < 4; ++kk) {
        v16bf bf = make_frag(&Kl[(nt * 16 + lm) * KSTR + kk * 32 + laneHalf * 8]);
        c = __builtin_amdgcn_wmma_f32_16x16x32_bf16(false, qf[kk], false, bf,
                                                    (short)0, c, false, false);
      }
      sfrag[nt] = c;
    }

    // ---- online softmax (rows on 16-lane groups; shfl_xor reductions) ----
    float bm[8];
#pragma unroll
    for (int v = 0; v < 8; ++v) {
      float x = sfrag[0][v];
      x = fmaxf(x, sfrag[1][v]); x = fmaxf(x, sfrag[2][v]); x = fmaxf(x, sfrag[3][v]);
      bm[v] = x;
    }
#pragma unroll
    for (int mk = 8; mk >= 1; mk >>= 1)
#pragma unroll
      for (int v = 0; v < 8; ++v)
        bm[v] = fmaxf(bm[v], __shfl_xor(bm[v], mk, 32));

    float mnew[8], alpha[8], rs[8];
#pragma unroll
    for (int v = 0; v < 8; ++v) {
      mnew[v]  = fmaxf(m_i[v], bm[v]);
      alpha[v] = __expf(m_i[v] - mnew[v]);
      rs[v]    = 0.f;
    }

    __bf16* pw = &Pl[wave * 16 * PSTR];
#pragma unroll
    for (int nt = 0; nt < 4; ++nt)
#pragma unroll
      for (int v = 0; v < 8; ++v) {
        const float p = __expf(sfrag[nt][v] - mnew[v]);
        rs[v] += p;
        pw[(v + 8 * laneHalf) * PSTR + nt * 16 + lm] = f2bf(p);
      }
#pragma unroll
    for (int mk = 8; mk >= 1; mk >>= 1)
#pragma unroll
      for (int v = 0; v < 8; ++v)
        rs[v] += __shfl_xor(rs[v], mk, 32);

#pragma unroll
    for (int v = 0; v < 8; ++v) {
      l_i[v] = l_i[v] * alpha[v] + rs[v];
      m_i[v] = mnew[v];
    }
#pragma unroll
    for (int dt = 0; dt < 8; ++dt)
#pragma unroll
      for (int v = 0; v < 8; ++v)
        acc[dt][v] *= alpha[v];

    // ---- O += P . V ------------------------------------------------------
    v16bf pf[2];
#pragma unroll
    for (int ks = 0; ks < 2; ++ks)
      pf[ks] = make_frag(&pw[lm * PSTR + ks * 32 + laneHalf * 8]);
#pragma unroll
    for (int dt = 0; dt < 8; ++dt)
#pragma unroll
      for (int ks = 0; ks < 2; ++ks) {
        v16bf bf = make_frag(&VTl[(dt * 16 + lm) * VSTR + ks * 32 + laneHalf * 8]);
        acc[dt] = __builtin_amdgcn_wmma_f32_16x16x32_bf16(false, pf[ks], false, bf,
                                                          (short)0, acc[dt], false, false);
      }
    __syncthreads();                       // tiles reused next iteration
  }

  // ---- reduce the 4 partial band-V column sums ----------------------------
  {
    const int d4 = (tid & 31) << 2;
    Vbp[tid >> 5][d4 + 0] = vb4[0];
    Vbp[tid >> 5][d4 + 1] = vb4[1];
    Vbp[tid >> 5][d4 + 2] = vb4[2];
    Vbp[tid >> 5][d4 + 3] = vb4[3];
  }
  __syncthreads();

  // ---- epilogue: fold out-of-band zero-score columns analytically ---------
  float sc[8], eo[8], lf[8];
#pragma unroll
  for (int v = 0; v < 8; ++v) {
    const float mfin = (n_out > 0) ? fmaxf(m_i[v], 0.f) : m_i[v];
    sc[v] = __expf(m_i[v] - mfin);
    eo[v] = (n_out > 0) ? __expf(-mfin) : 0.f;
    lf[v] = l_i[v] * sc[v] + (float)n_out * eo[v];
  }
#pragma unroll
  for (int dt = 0; dt < 8; ++dt) {
    const int dim = dt * 16 + lm;
    const float vband = Vbp[0][dim] + Vbp[1][dim] + Vbp[2][dim] + Vbp[3][dim];
    const float vo = vsum[b * D_ + dim] - vband;
#pragma unroll
    for (int v = 0; v < 8; ++v) {
      const int row = q0 + v + 8 * laneHalf;
      const float o = acc[dt][v] * sc[v] + eo[v] * vo;
      out[((size_t)b * S_ + row) * D_ + dim] = o / lf[v];
    }
  }
}

// ---- host launch ----------------------------------------------------------
extern "C" void kernel_launch(void* const* d_in, const int* in_sizes, int n_in,
                              void* d_out, int out_size, void* d_ws, size_t ws_size,
                              hipStream_t stream) {
  const float* Q = (const float*)d_in[0];
  const float* K = (const float*)d_in[1];
  const float* V = (const float*)d_in[2];
  float* vsum = (float*)d_ws;                    // B_*D_ floats = 2 KB
  float* out  = (float*)d_out;

  vsum_kernel<<<dim3(B_), 128, 0, stream>>>(V, vsum);

  dim3 grid(NB, B_);
  SparseAttention_59399397704373_kernel<<<grid, 128, 0, stream>>>(Q, K, V, vsum, out);
}
